// GAT_19645180412625
// MI455X (gfx1250) — matrix operand order repaired and checked by reading, not compiled, not run on hardware
//
#include <hip/hip_runtime.h>
#include <hip/hip_bf16.h>
#include <math.h>

typedef __attribute__((ext_vector_type(16))) _Float16 v16h;
typedef __attribute__((ext_vector_type(8)))  _Float16 v8h;
typedef __attribute__((ext_vector_type(8)))  float    v8f;

__device__ __forceinline__ float lrelu(float a) { return a > 0.f ? a : 0.2f * a; }

// Ordered-int float atomic max (works for all finite floats).
__device__ __forceinline__ void atomicMaxF(float* addr, float v) {
  int vi = __float_as_int(v);
  if (vi >= 0) atomicMax((int*)addr, vi);
  else         atomicMin((unsigned int*)addr, (unsigned int)vi);
}

// ---------------- one-shot weight packing (f16, B-fragment order) ----------------
// W2lT/W2rT: [col 0..31][K 0..63]  (transposed so a lane's 16 fragment halves are two contiguous 8-runs)
// W3T:       [col 0..31][K 0..31]  packed q|k|v|skip (8-col slots, col 7 of each slot = 0)
__global__ void pack_w(const float* __restrict__ W2l, const float* __restrict__ W2r,
                       const float* __restrict__ Wq, const float* __restrict__ Wk,
                       const float* __restrict__ Wv, const float* __restrict__ Wsk,
                       _Float16* __restrict__ W2lT, _Float16* __restrict__ W2rT,
                       _Float16* __restrict__ W3T) {
  int t = blockIdx.x * blockDim.x + threadIdx.x;
  if (t < 2048) {
    int col = t >> 6, K = t & 63;
    W2lT[t] = (_Float16)W2l[K * 32 + col];
    W2rT[t] = (_Float16)W2r[K * 32 + col];
  } else if (t < 3072) {
    int u = t - 2048;
    int col = u >> 5, K = u & 31;
    int which = col >> 3, sub = col & 7;
    const float* Wm[4] = {Wq, Wk, Wv, Wsk};
    W3T[u] = (sub < 7) ? (_Float16)Wm[which][K * 7 + sub] : (_Float16)0.f;
  }
}

// ---------------- Layer 1: GATConv(3,16,heads=4) ----------------

__global__ void l1_node(const float* __restrict__ x, const float* __restrict__ W1,
                        const float* __restrict__ as1, const float* __restrict__ ad1,
                        float* __restrict__ h1, float* __restrict__ asrc,
                        float* __restrict__ adst, int N) {
  int n = blockIdx.x * blockDim.x + threadIdx.x;
  if (n >= N) return;
  float x0 = x[n * 3 + 0], x1 = x[n * 3 + 1], x2 = x[n * 3 + 2];
  float sa[4] = {0.f, 0.f, 0.f, 0.f}, sd[4] = {0.f, 0.f, 0.f, 0.f};
#pragma unroll
  for (int j = 0; j < 64; ++j) {
    float h = x0 * W1[j] + x1 * W1[64 + j] + x2 * W1[128 + j];
    h1[(size_t)n * 64 + j] = h;
    int hh = j >> 4;
    sa[hh] += h * as1[j];
    sd[hh] += h * ad1[j];
  }
#pragma unroll
  for (int hh = 0; hh < 4; ++hh) {
    asrc[n * 4 + hh] = sa[hh];
    adst[n * 4 + hh] = sd[hh];
  }
}

__global__ void l1_edge_max(const int* __restrict__ ei, const float* __restrict__ asrc,
                            const float* __restrict__ adst, float* __restrict__ amax,
                            int E, int N) {
  int e = blockIdx.x * blockDim.x + threadIdx.x;
  if (e >= E + N) return;
  int s, d;
  if (e < E) { s = ei[e]; d = ei[E + e]; } else { s = d = e - E; }
#pragma unroll
  for (int h = 0; h < 4; ++h) {
    float a = lrelu(asrc[s * 4 + h] + adst[d * 4 + h]);
    atomicMaxF(&amax[d * 4 + h], a);
  }
}

// One wave per edge; lane l covers channels 2l, 2l+1 (head = l>>3).
// Source-row gather and dst-row scatter are fully coalesced (2 cachelines each).
__global__ void l1_edge_acc(const int* __restrict__ ei, const float* __restrict__ asrc,
                            const float* __restrict__ adst, const float* __restrict__ amax,
                            const float* __restrict__ h1, float* __restrict__ denom,
                            float* __restrict__ out1, int E, int N) {
  int wv   = (blockIdx.x * blockDim.x + threadIdx.x) >> 5;
  int lane = threadIdx.x & 31;
  if (wv >= E + N) return;
  int s, d;
  if (wv < E) { s = ei[wv]; d = ei[E + wv]; } else { s = d = wv - E; }
  int h = lane >> 3;
  float a  = lrelu(asrc[s * 4 + h] + adst[d * 4 + h]);
  float ex = __expf(a - amax[d * 4 + h]);
  if ((lane & 7) == 0) atomicAdd(&denom[d * 4 + h], ex);
  const float2 hv = *(const float2*)(h1 + (size_t)s * 64 + 2 * lane);
  float* od = out1 + (size_t)d * 64 + 2 * lane;
  atomicAdd(&od[0], hv.x * ex);
  atomicAdd(&od[1], hv.y * ex);
}

// x1 = relu(out1/denom + b1), stored as f16 for WMMA
__global__ void l1_final(const float* __restrict__ out1, const float* __restrict__ denom,
                         const float* __restrict__ b1, _Float16* __restrict__ x1h, int N) {
  int i = blockIdx.x * blockDim.x + threadIdx.x;
  if (i >= N * 64) return;
  int n = i >> 6, j = i & 63, h = j >> 4;
  float v = out1[i] / (denom[n * 4 + h] + 1e-16f) + b1[j];
  x1h[i] = (_Float16)fmaxf(v, 0.f);
}

// ---------------- Layer 2: GATv2Conv(64,16,heads=2): WMMA hl/hr ----------------

__device__ __forceinline__ v16h frag_from(const _Float16* p0, const _Float16* p1) {
  v8h lo = *(const v8h*)p0;
  v8h hi = *(const v8h*)p1;
  v16h r;
#pragma unroll
  for (int i = 0; i < 8; ++i) { r[i] = lo[i]; r[i + 8] = hi[i]; }
  return r;
}

__global__ void l2_wmma(const _Float16* __restrict__ x1h,
                        const _Float16* __restrict__ W2lT, const float* __restrict__ b2l,
                        const _Float16* __restrict__ W2rT, const float* __restrict__ b2r,
                        float* __restrict__ hl, float* __restrict__ hr, int N) {
  int wave = threadIdx.x >> 5, lane = threadIdx.x & 31;
  int tile = blockIdx.x * (blockDim.x >> 5) + wave;
  int row0 = tile * 16;
  if (row0 >= N) return;                      // wave-uniform exit, EXEC stays full
  int g = lane >> 4, nl = lane & 15;
  int arow = row0 + nl; if (arow >= N) arow = N - 1;

  v8f zero = {};
  v8f accl[2] = {zero, zero};
  v8f accr[2] = {zero, zero};

#pragma unroll
  for (int kk = 0; kk < 2; ++kk) {
    // A 16x32 f16: M = lane&15; elems 0-7 -> K=kk*32+8g.., elems 8-15 -> K=kk*32+16+8g..
    const _Float16* ap = x1h + (size_t)arow * 64 + kk * 32 + g * 8;
    v16h a = frag_from(ap, ap + 16);
#pragma unroll
    for (int nb = 0; nb < 2; ++nb) {
      int col = nb * 16 + nl;
      // B 32x16 f16: elems 0-7 -> K=8g.., elems 8-15 -> K=16+8g..; from W*T[col][K]
      const _Float16* blp = W2lT + col * 64 + kk * 32 + g * 8;
      const _Float16* brp = W2rT + col * 64 + kk * 32 + g * 8;
      v16h bl = frag_from(blp, blp + 16);
      v16h br = frag_from(brp, brp + 16);
      accl[nb] = __builtin_amdgcn_wmma_f32_16x16x32_f16(false, a, false, bl,
                                                        (short)0, accl[nb], false, false);
      accr[nb] = __builtin_amdgcn_wmma_f32_16x16x32_f16(false, a, false, br,
                                                        (short)0, accr[nb], false, false);
    }
  }

#pragma unroll
  for (int nb = 0; nb < 2; ++nb) {
    int col = nb * 16 + nl;
    float bl_ = b2l[col], br_ = b2r[col];
#pragma unroll
    for (int r = 0; r < 8; ++r) {
      int m = row0 + r + g * 8;               // C/D: vgpr r -> M=r (+8 for hi half-wave)
      if (m < N) {
        hl[(size_t)m * 32 + col] = accl[nb][r] + bl_;
        hr[(size_t)m * 32 + col] = accr[nb][r] + br_;
      }
    }
  }
}

// One wave per edge; lane = channel (head = lane>>4); butterfly-reduce the
// 16-channel attention dot within each half-wave.
__global__ void l2_edge_max(const int* __restrict__ ei, const float* __restrict__ hl,
                            const float* __restrict__ hr, const float* __restrict__ att2,
                            float* __restrict__ amax2, int E, int N) {
  int wv   = (blockIdx.x * blockDim.x + threadIdx.x) >> 5;
  int lane = threadIdx.x & 31;
  if (wv >= E + N) return;
  int s, d;
  if (wv < E) { s = ei[wv]; d = ei[E + wv]; } else { s = d = wv - E; }
  float hls = hl[(size_t)s * 32 + lane];
  float hrd = hr[(size_t)d * 32 + lane];
  float t = lrelu(hls + hrd) * att2[lane];
  t += __shfl_xor(t, 1, 16);
  t += __shfl_xor(t, 2, 16);
  t += __shfl_xor(t, 4, 16);
  t += __shfl_xor(t, 8, 16);
  if ((lane & 15) == 0) atomicMaxF(&amax2[d * 2 + (lane >> 4)], t);
}

__global__ void l2_edge_acc(const int* __restrict__ ei, const float* __restrict__ hl,
                            const float* __restrict__ hr, const float* __restrict__ att2,
                            const float* __restrict__ amax2, float* __restrict__ denom2,
                            float* __restrict__ out2, int E, int N) {
  int wv   = (blockIdx.x * blockDim.x + threadIdx.x) >> 5;
  int lane = threadIdx.x & 31;
  if (wv >= E + N) return;
  int s, d;
  if (wv < E) { s = ei[wv]; d = ei[E + wv]; } else { s = d = wv - E; }
  int h = lane >> 4;
  float hls = hl[(size_t)s * 32 + lane];
  float hrd = hr[(size_t)d * 32 + lane];
  float t = lrelu(hls + hrd) * att2[lane];
  t += __shfl_xor(t, 1, 16);
  t += __shfl_xor(t, 2, 16);
  t += __shfl_xor(t, 4, 16);
  t += __shfl_xor(t, 8, 16);
  float ex = __expf(t - amax2[d * 2 + h]);
  if ((lane & 15) == 0) atomicAdd(&denom2[d * 2 + h], ex);
  atomicAdd(&out2[(size_t)d * 32 + lane], hls * ex);
}

// x2 = out2/denom + b2, stored f16 for layer-3 WMMA
__global__ void l2_final(const float* __restrict__ out2, const float* __restrict__ denom2,
                         const float* __restrict__ b2, _Float16* __restrict__ x2h, int N) {
  int i = blockIdx.x * blockDim.x + threadIdx.x;
  if (i >= N * 32) return;
  int n = i >> 5, j = i & 31, h = j >> 4;
  x2h[i] = (_Float16)(out2[i] / (denom2[n * 2 + h] + 1e-16f) + b2[j]);
}

// ---------------- Layer 3: TransformerConv(32,7): WMMA packed q|k|v|skip ----------------
// P[n, 0:7]=q  P[n, 8:15]=k  P[n,16:23]=v  P[n,24:31]=skip

__global__ void l3_wmma(const _Float16* __restrict__ x2h, const _Float16* __restrict__ W3T,
                        const float* __restrict__ bq, const float* __restrict__ bk,
                        const float* __restrict__ bv, const float* __restrict__ bsk,
                        float* __restrict__ P, int N) {
  int wave = threadIdx.x >> 5, lane = threadIdx.x & 31;
  int tile = blockIdx.x * (blockDim.x >> 5) + wave;
  int row0 = tile * 16;
  if (row0 >= N) return;
  int g = lane >> 4, nl = lane & 15;
  int arow = row0 + nl; if (arow >= N) arow = N - 1;

  const _Float16* ap = x2h + (size_t)arow * 32 + g * 8;
  v16h a = frag_from(ap, ap + 16);

  const float* Bm[4] = {bq, bk, bv, bsk};
  v8f zero = {};
  v8f acc[2] = {zero, zero};

#pragma unroll
  for (int nb = 0; nb < 2; ++nb) {
    int col = nb * 16 + nl;
    const _Float16* bp = W3T + col * 32 + g * 8;
    v16h b = frag_from(bp, bp + 16);
    acc[nb] = __builtin_amdgcn_wmma_f32_16x16x32_f16(false, a, false, b,
                                                     (short)0, acc[nb], false, false);
  }

#pragma unroll
  for (int nb = 0; nb < 2; ++nb) {
    int col = nb * 16 + nl;
    int which = col >> 3, sub = col & 7;
    float bias = (sub < 7) ? Bm[which][sub] : 0.f;
#pragma unroll
    for (int r = 0; r < 8; ++r) {
      int m = row0 + r + g * 8;
      if (m < N) P[(size_t)m * 32 + col] = acc[nb][r] + bias;
    }
  }
}

__global__ void l3_edge_max(const int* __restrict__ ei, const float* __restrict__ P,
                            float* __restrict__ amax3, int E) {
  int e = blockIdx.x * blockDim.x + threadIdx.x;
  if (e >= E) return;
  int s = ei[e], d = ei[E + e];
  const float* q = P + (size_t)d * 32;
  const float* k = P + (size_t)s * 32 + 8;
  float acc = 0.f;
#pragma unroll
  for (int c = 0; c < 7; ++c) acc += q[c] * k[c];
  acc *= 0.3779644730092272f;  // 1/sqrt(7)
  atomicMaxF(&amax3[d], acc);
}

__global__ void l3_edge_acc(const int* __restrict__ ei, const float* __restrict__ P,
                            const float* __restrict__ amax3, float* __restrict__ denom3,
                            float* __restrict__ dout, int E) {
  int e = blockIdx.x * blockDim.x + threadIdx.x;
  if (e >= E) return;
  int s = ei[e], d = ei[E + e];
  const float* q = P + (size_t)d * 32;
  const float* k = P + (size_t)s * 32 + 8;
  const float* v = P + (size_t)s * 32 + 16;
  __builtin_prefetch(v, 0, 0);   // global_prefetch_b8
  float acc = 0.f;
#pragma unroll
  for (int c = 0; c < 7; ++c) acc += q[c] * k[c];
  acc *= 0.3779644730092272f;
  float ex = __expf(acc - amax3[d]);
  atomicAdd(&denom3[d], ex);
#pragma unroll
  for (int c = 0; c < 7; ++c)
    atomicAdd(&dout[(size_t)d * 7 + c], v[c] * ex);
}

__global__ void l3_final(float* __restrict__ dout, const float* __restrict__ denom3,
                         const float* __restrict__ P, int N) {
  int i = blockIdx.x * blockDim.x + threadIdx.x;
  if (i >= N * 7) return;
  int n = i / 7, c = i - n * 7;
  dout[i] = dout[i] / (denom3[n] + 1e-16f) + P[(size_t)n * 32 + 24 + c];
}

// ---------------- host side ----------------

extern "C" void kernel_launch(void* const* d_in, const int* in_sizes, int n_in,
                              void* d_out, int out_size, void* d_ws, size_t ws_size,
                              hipStream_t stream) {
  (void)n_in; (void)ws_size;
  const float* x   = (const float*)d_in[0];
  const int*   ei  = (const int*)d_in[1];
  const float* W1  = (const float*)d_in[2];
  const float* as1 = (const float*)d_in[3];
  const float* ad1 = (const float*)d_in[4];
  const float* b1  = (const float*)d_in[5];
  const float* W2l = (const float*)d_in[6];
  const float* b2l = (const float*)d_in[7];
  const float* W2r = (const float*)d_in[8];
  const float* b2r = (const float*)d_in[9];
  const float* at2 = (const float*)d_in[10];
  const float* b2  = (const float*)d_in[11];
  const float* Wq  = (const float*)d_in[12];
  const float* bq  = (const float*)d_in[13];
  const float* Wk  = (const float*)d_in[14];
  const float* bk  = (const float*)d_in[15];
  const float* Wv  = (const float*)d_in[16];
  const float* bv  = (const float*)d_in[17];
  const float* Wsk = (const float*)d_in[18];
  const float* bsk = (const float*)d_in[19];
  float* dout = (float*)d_out;

  int N = in_sizes[0] / 3;
  int E = in_sizes[1] / 2;
  size_t fN = (size_t)N;
  char* ws = (char*)d_ws;

  // Workspace layout (byte offsets), with reuse across layers:
  //  [0,    256N)  h1 f32[N,64]   -> later: x1h f16[N,64] | x2h f16[N,32]
  //  [128N, 256N)  hl f32[N,32]   -> later: P f32[N,32]
  //  [256N, 512N)  out1 f32[N,64] -> later: hr @256N, out2 @384N
  //  [512N..576N)  asrc1 | adst1 | amax | denom  (f32[N,4] each)
  //  [576N..)      packed f16 weights (10 KB)
  float*    h1    = (float*)(ws);
  float*    out1  = (float*)(ws + fN * 256);
  float*    asrc  = (float*)(ws + fN * 512);
  float*    adst  = (float*)(ws + fN * 528);
  float*    amax  = (float*)(ws + fN * 544);
  float*    denom = (float*)(ws + fN * 560);
  _Float16* x1h   = (_Float16*)(ws);
  float*    hl    = (float*)(ws + fN * 128);
  float*    hr    = (float*)(ws + fN * 256);
  float*    out2  = (float*)(ws + fN * 384);
  _Float16* x2h   = (_Float16*)(ws);
  float*    P     = (float*)(ws + fN * 128);
  _Float16* W2lT  = (_Float16*)(ws + fN * 576);
  _Float16* W2rT  = (_Float16*)(ws + fN * 576 + 4096);
  _Float16* W3T   = (_Float16*)(ws + fN * 576 + 8192);

  const int B = 256;
  int gN   = (N + B - 1) / B;
  int gEN  = (E + N + B - 1) / B;          // thread-per-edge
  int gENw = (E + N + 7) / 8;              // wave-per-edge (8 waves/block)
  int gE   = (E + B - 1) / B;
  int gW   = (N + 16 * 8 - 1) / (16 * 8);  // 8 waves/block, 16 rows/wave

  pack_w<<<12, B, 0, stream>>>(W2l, W2r, Wq, Wk, Wv, Wsk, W2lT, W2rT, W3T);

  // ---- layer 1 ----
  hipMemsetAsync(amax, 0xFF, fN * 16, stream);   // NaN bits; overwritten by pass 1
  hipMemsetAsync(denom, 0, fN * 16, stream);
  hipMemsetAsync(out1, 0, fN * 256, stream);
  l1_node<<<gN, B, 0, stream>>>(x, W1, as1, ad1, h1, asrc, adst, N);
  l1_edge_max<<<gEN, B, 0, stream>>>(ei, asrc, adst, amax, E, N);
  l1_edge_acc<<<gENw, B, 0, stream>>>(ei, asrc, adst, amax, h1, denom, out1, E, N);
  l1_final<<<(N * 64 + B - 1) / B, B, 0, stream>>>(out1, denom, b1, x1h, N);

  // ---- layer 2 ----
  hipMemsetAsync(amax, 0xFF, fN * 8, stream);
  hipMemsetAsync(denom, 0, fN * 8, stream);
  hipMemsetAsync(out2, 0, fN * 128, stream);
  l2_wmma<<<gW, B, 0, stream>>>(x1h, W2lT, b2l, W2rT, b2r, hl, hr, N);
  l2_edge_max<<<gENw, B, 0, stream>>>(ei, hl, hr, at2, amax, E, N);
  l2_edge_acc<<<gENw, B, 0, stream>>>(ei, hl, hr, at2, amax, denom, out2, E, N);
  l2_final<<<(N * 32 + B - 1) / B, B, 0, stream>>>(out2, denom, b2, x2h, N);

  // ---- layer 3 ----
  hipMemsetAsync(amax, 0xFF, fN * 4, stream);
  hipMemsetAsync(denom, 0, fN * 4, stream);
  hipMemsetAsync(dout, 0, (size_t)out_size * sizeof(float), stream);
  l3_wmma<<<gW, B, 0, stream>>>(x2h, W3T, bq, bk, bv, bsk, P, N);
  l3_edge_max<<<gE, B, 0, stream>>>(ei, P, amax, E);
  l3_edge_acc<<<gE, B, 0, stream>>>(ei, P, amax, denom, dout, E);
  l3_final<<<(N * 7 + B - 1) / B, B, 0, stream>>>(dout, denom, P, N);
}
